// AttentionBottleNeck_26259430047848
// MI455X (gfx1250) — matrix operand order repaired and checked by reading, not compiled, not run on hardware
//
#include <hip/hip_runtime.h>
#include <hip/hip_bf16.h>

typedef __attribute__((ext_vector_type(16))) _Float16 v16h;
typedef __attribute__((ext_vector_type(8)))  _Float16 v8h;
typedef __attribute__((ext_vector_type(8)))  float    v8f;

#define WMMA_F32_F16(a, b, c) \
  __builtin_amdgcn_wmma_f32_16x16x32_f16(false, (a), false, (b), (short)0, (c), false, false)

constexpr int kB = 64, kC = 256, kL = 4096, kHeads = 8, kQ = 16, kFH = 512;
constexpr size_t kCL = (size_t)kC * kL;
constexpr int kTpad = 264;  // LN transpose tile ld (halves), multiple of 8 for b128 align

// ---------------- WMMA fragment helpers (CDNA5 16x16x32 f16 layouts) ----------------

// A matrix 16x32 (MxK), row-major source with leading dim `ld` (halves).
// lane<16: K=0..7 & 16..23 of row m=lane ; lane>=16: K=8..15 & 24..31 of row m=lane-16.
__device__ inline v16h load_a_frag(const _Float16* p, int ld) {
  const int lane = threadIdx.x & 31;
  const int m = lane & 15;
  const int kb = (lane < 16) ? 0 : 8;
  const _Float16* row = p + (size_t)m * ld + kb;
  v8h lo = *(const v8h*)(row);
  v8h hi = *(const v8h*)(row + 16);
  v16h a;
#pragma unroll
  for (int i = 0; i < 8; ++i) { a[i] = lo[i]; a[i + 8] = hi[i]; }
  return a;
}

// B matrix 32x16 (KxN) where B[k][n] = p[n*ld + k] (each column contiguous in K).
// lane<16: col n=lane, K=0..7 & 16..23 ; lane>=16: col n=lane-16, K=8..15 & 24..31.
__device__ inline v16h load_b_frag_kcontig(const _Float16* p, int ld) {
  const int lane = threadIdx.x & 31;
  const int n = lane & 15;
  const int kb = (lane < 16) ? 0 : 8;
  const _Float16* col = p + (size_t)n * ld + kb;
  v8h lo = *(const v8h*)(col);
  v8h hi = *(const v8h*)(col + 16);
  v16h b;
#pragma unroll
  for (int i = 0; i < 8; ++i) { b[i] = lo[i]; b[i + 8] = hi[i]; }
  return b;
}

// ---------------- CDNA5 async global->LDS DMA (ASYNCcnt-tracked) ----------------
__device__ inline void async_copy16_g2l(_Float16* lds_dst, const _Float16* gsrc) {
  unsigned lds_addr = (unsigned)(unsigned long long)lds_dst;  // LDS aperture: addr[31:0]
  asm volatile("global_load_async_to_lds_b128 %0, %1, off"
               :: "v"(lds_addr), "v"(gsrc)
               : "memory");
}
__device__ inline void wait_async0() {
  asm volatile("s_wait_asynccnt 0x0" ::: "memory");
}

// ---------------- Kernel 0: f32 -> f16 weight conversion ----------------
__global__ void cvt_f32_f16(const float* __restrict__ src, _Float16* __restrict__ dst, int n) {
  int i = blockIdx.x * blockDim.x + threadIdx.x;
  if (i < n) dst[i] = (_Float16)src[i];
}

// ---------------- Kernel 1: LayerNorm over C, emit TRANSPOSED f16: xn_T[b][l][c] -------------
// Thread-per-l for coalesced channel-strided reads; 132 KB LDS tile re-transposes so the
// [l][c] global writes are fully coalesced. Downstream GEMMs then get K-contiguous B
// operands directly from global memory (no LDS transpose staging anywhere else).
__global__ __launch_bounds__(256) void layernorm_k(const float* __restrict__ x,
                                                   const float* __restrict__ gamma,
                                                   const float* __restrict__ beta,
                                                   _Float16* __restrict__ xnT) {
  extern __shared__ _Float16 Tls[];  // [256 l][kTpad c]
  __shared__ float sg[kC], sb[kC];
  const int tid = threadIdx.x;
  sg[tid] = gamma[tid];
  sb[tid] = beta[tid];
  __syncthreads();
  const int b = blockIdx.x >> 4;
  const int l0 = (blockIdx.x & 15) << 8;
  const float* px = x + (size_t)b * kCL + l0 + tid;  // lanes coalesced over l
  float s = 0.f, s2 = 0.f;
#pragma unroll 4
  for (int c = 0; c < kC; ++c) {
    float v = px[(size_t)c * kL];
    s += v; s2 += v * v;
  }
  const float mu = s * (1.0f / kC);
  const float var = s2 * (1.0f / kC) - mu * mu;
  const float rinv = rsqrtf(var + 1e-6f);
#pragma unroll 4
  for (int c = 0; c < kC; ++c) {
    float v = px[(size_t)c * kL];
    Tls[tid * kTpad + c] = (_Float16)((v - mu) * rinv * sg[c] + sb[c]);
  }
  __syncthreads();
  // coalesced [l][c] writes: 8-thread groups emit 512B rows
  _Float16* po = xnT + (size_t)b * kCL + (size_t)l0 * kC;
  const int rr = tid >> 3;        // 0..31
  const int cc = (tid & 7) << 5;  // 0,32,..,224
  for (int g = 0; g < 8; ++g) {
    const int row = g * 32 + rr;
#pragma unroll
    for (int u = 0; u < 4; ++u) {
      v8h d = *(const v8h*)(&Tls[row * kTpad + cc + u * 8]);
      *(v8h*)(po + (size_t)row * kC + cc + u * 8) = d;
    }
  }
}

// ---------------- Kernel 2: fused grouped-conv logits (WMMA) + row softmax ----------------
// One block per (b,h). B fragments come straight from xn_T (K-contiguous b128 loads);
// the 16x4096 logit tile lives in 128 KB LDS, so logits never touch HBM. No barriers
// in the WMMA phase; a single barrier before the LDS-resident softmax.
__global__ __launch_bounds__(256) void logits_softmax_k(const _Float16* __restrict__ xnT,
                                                        const _Float16* __restrict__ attnw,
                                                        _Float16* __restrict__ aw) {
  extern __shared__ _Float16 Sl[];  // [16 q][4096 l]
  __shared__ float red[256];

  const int bh = blockIdx.x;  // 0..511
  const int b = bh >> 3, h = bh & 7;
  const int tid = threadIdx.x;
  const int w = tid >> 5;
  const v16h a = load_a_frag(attnw + h * (kQ * 32), 32);  // 16x32 weights, resident
  const _Float16* xbase = xnT + (size_t)b * kCL + h * 32;  // [l][c] rows, head slice
  const int lane = tid & 31;
  const int nn = lane & 15;
  const int ro = (lane < 16) ? 0 : 8;

  for (int st = w; st < kL / 16; st += 8) {  // 256 l-subtiles, wave-strided
    const int l0 = st << 4;
    v16h bf = load_b_frag_kcontig(xbase + (size_t)l0 * kC, kC);
    v8f c = {};
    c = WMMA_F32_F16(a, bf, c);
#pragma unroll
    for (int r = 0; r < 8; ++r)
      Sl[(r + ro) * kL + l0 + nn] = (_Float16)c[r];
  }
  __syncthreads();

  // row-wise softmax over L=4096 straight out of LDS
  _Float16* obase = aw + ((size_t)bh * kQ) * kL;
  for (int q = 0; q < kQ; ++q) {
    const _Float16* row = Sl + q * kL;
    float mx = -3.4e38f;
#pragma unroll
    for (int i = 0; i < 16; ++i) mx = fmaxf(mx, (float)row[tid + (i << 8)]);
    red[tid] = mx; __syncthreads();
    for (int s2 = 128; s2 > 0; s2 >>= 1) {
      if (tid < s2) red[tid] = fmaxf(red[tid], red[tid + s2]);
      __syncthreads();
    }
    mx = red[0]; __syncthreads();
    float ev[16];
    float sum = 0.f;
#pragma unroll
    for (int i = 0; i < 16; ++i) {
      ev[i] = __expf((float)row[tid + (i << 8)] - mx);
      sum += ev[i];
    }
    red[tid] = sum; __syncthreads();
    for (int s2 = 128; s2 > 0; s2 >>= 1) {
      if (tid < s2) red[tid] += red[tid + s2];
      __syncthreads();
    }
    const float inv = 1.0f / red[0];
    __syncthreads();  // red[] reused next q
#pragma unroll
    for (int i = 0; i < 16; ++i)
      obase[(size_t)q * kL + tid + (i << 8)] = (_Float16)(ev[i] * inv);
  }
}

// ---------------- Kernel 3: V = val_w @ xn + b (per-batch 256x4096x256 GEMM) ----------------
// Block tile: 64(e) x 128(l); 8 waves in 2x4; each wave 2x2 WMMA tiles.
// A tile staged via CDNA5 async global->LDS DMA; B fragments direct from xn_T (b128).
__global__ __launch_bounds__(256) void value_gemm_k(const _Float16* __restrict__ xnT,
                                                    const _Float16* __restrict__ vw,
                                                    const float* __restrict__ vb,
                                                    _Float16* __restrict__ V) {
  __shared__ _Float16 At[64 * 40];  // [e][k], padded ld=40
  const int bx = blockIdx.x;
  const int lt = bx & 31, et = (bx >> 5) & 3, b = bx >> 7;
  const int e0 = et * 64, l0 = lt * 128;
  const int tid = threadIdx.x;
  const int w = tid >> 5;
  const int wm = w >> 2, wn = w & 3;
  v8f acc[2][2] = {};
  const _Float16* xb = xnT + (size_t)b * kCL;

  for (int k0 = 0; k0 < kC; k0 += 32) {
    __syncthreads();
    {  // A tile: 64 rows x 32 k -> async DMA, 16B per lane
      const int r = tid >> 2, cg = (tid & 3) << 3;
      async_copy16_g2l(&At[r * 40 + cg], vw + (size_t)(e0 + r) * kC + k0 + cg);
    }
    wait_async0();
    __syncthreads();
    v16h bf[2];
#pragma unroll
    for (int ns = 0; ns < 2; ++ns)
      bf[ns] = load_b_frag_kcontig(xb + (size_t)(l0 + wn * 32 + ns * 16) * kC + k0, kC);
#pragma unroll
    for (int ms = 0; ms < 2; ++ms) {
      v16h a = load_a_frag(&At[(wm * 32 + ms * 16) * 40], 40);
#pragma unroll
      for (int ns = 0; ns < 2; ++ns)
        acc[ms][ns] = WMMA_F32_F16(a, bf[ns], acc[ms][ns]);
    }
  }
  const int lane = tid & 31;
  const int nn = lane & 15, ro = (lane < 16) ? 0 : 8;
  _Float16* vbase = V + (size_t)b * kCL;
#pragma unroll
  for (int ms = 0; ms < 2; ++ms)
#pragma unroll
    for (int ns = 0; ns < 2; ++ns)
#pragma unroll
      for (int r = 0; r < 8; ++r) {
        const int e = e0 + wm * 32 + ms * 16 + ro + r;
        const int l = l0 + wn * 32 + ns * 16 + nn;
        vbase[(size_t)e * kL + l] = (_Float16)(acc[ms][ns][r] + vb[e]);
      }
}

// ---------------- Kernel 4: A[q,d] = sum_l Aw[q,l] * V[d,l]  (16x32, K=4096) ----------------
__global__ __launch_bounds__(256) void pool_k(const _Float16* __restrict__ aw,
                                              const _Float16* __restrict__ V,
                                              _Float16* __restrict__ aflat) {
  const int idx = blockIdx.x * 8 + (threadIdx.x >> 5);  // (b,h) task, 0..511
  const int b = idx >> 3, h = idx & 7;
  const _Float16* arow = aw + ((size_t)idx * kQ) * kL;
  const _Float16* vbase = V + (size_t)b * kCL + (size_t)(h * 32) * kL;
  v8f acc[2] = {};
  for (int l0 = 0; l0 < kL; l0 += 32) {
    if (l0 + 1024 < kL) {  // global_prefetch_b8 ahead of the stream
      __builtin_prefetch(arow + l0 + 1024, 0, 1);
      __builtin_prefetch(vbase + l0 + 1024, 0, 1);
    }
    v16h a = load_a_frag(arow + l0, kL);
#pragma unroll
    for (int ns = 0; ns < 2; ++ns) {
      v16h bf = load_b_frag_kcontig(vbase + (size_t)(ns * 16) * kL + l0, kL);
      acc[ns] = WMMA_F32_F16(a, bf, acc[ns]);
    }
  }
  const int lane = threadIdx.x & 31;
  const int nn = lane & 15, ro = (lane < 16) ? 0 : 8;
#pragma unroll
  for (int ns = 0; ns < 2; ++ns)
#pragma unroll
    for (int r = 0; r < 8; ++r) {
      const int q = ro + r;
      const int dd = ns * 16 + nn;
      // torch reshape order (h, q, d)
      aflat[(size_t)b * 4096 + h * 512 + q * 32 + dd] = (_Float16)acc[ns][r];
    }
}

// ---------------- Kernel 5: out = Aflat @ fin_w^T + fin_b (64x512x4096) ----------------
__global__ __launch_bounds__(256) void final_gemm_k(const _Float16* __restrict__ aflat,
                                                    const _Float16* __restrict__ fw,
                                                    const float* __restrict__ fb,
                                                    float* __restrict__ out) {
  const int idx = blockIdx.x * 8 + (threadIdx.x >> 5);  // output tile, 0..127
  const int mt = idx >> 5, nt = idx & 31;
  v8f acc = {};
  const _Float16* abase = aflat + (size_t)(mt * 16) * 4096;
  const _Float16* bbase = fw + (size_t)(nt * 16) * 4096;  // fin_w[f][k]: K-contiguous
  for (int k0 = 0; k0 < 4096; k0 += 32) {
    if (k0 + 1024 < 4096) __builtin_prefetch(bbase + k0 + 1024, 0, 1);
    v16h a = load_a_frag(abase + k0, 4096);
    v16h bf = load_b_frag_kcontig(bbase + k0, 4096);
    acc = WMMA_F32_F16(a, bf, acc);
  }
  const int lane = threadIdx.x & 31;
  const int nn = lane & 15, ro = (lane < 16) ? 0 : 8;
  const int f = nt * 16 + nn;
  const float bias = fb[f];
#pragma unroll
  for (int r = 0; r < 8; ++r) {
    const int brow = mt * 16 + ro + r;
    out[(size_t)brow * kFH + f] = acc[r] + bias;
  }
}

extern "C" void kernel_launch(void* const* d_in, const int* in_sizes, int n_in,
                              void* d_out, int out_size, void* d_ws, size_t ws_size,
                              hipStream_t stream) {
  (void)in_sizes; (void)n_in; (void)out_size; (void)ws_size;
  const float* x     = (const float*)d_in[0];
  const float* g     = (const float*)d_in[1];
  const float* be    = (const float*)d_in[2];
  const float* attnw = (const float*)d_in[3];
  const float* valw  = (const float*)d_in[4];
  const float* valb  = (const float*)d_in[5];
  const float* finw  = (const float*)d_in[6];
  const float* finb  = (const float*)d_in[7];
  float* out = (float*)d_out;

  char* ws = (char*)d_ws;
  size_t off = 0;
  _Float16* xnT16   = (_Float16*)(ws + off); off += (size_t)kB * kCL * 2;                  // 134 MB
  _Float16* v16     = (_Float16*)(ws + off); off += (size_t)kB * kCL * 2;                  // 134 MB
  _Float16* aw16    = (_Float16*)(ws + off); off += (size_t)kB * kHeads * kQ * kL * 2;     // 67 MB
  _Float16* afl16   = (_Float16*)(ws + off); off += (size_t)kB * 4096 * 2;
  _Float16* attnw16 = (_Float16*)(ws + off); off += 8192 * 2;
  _Float16* valw16  = (_Float16*)(ws + off); off += (size_t)kC * kC * 2;
  _Float16* finw16  = (_Float16*)(ws + off); off += (size_t)kFH * kC * kQ * 2;

  cvt_f32_f16<<<(4096 + 255) / 256, 256, 0, stream>>>(attnw, attnw16, 4096);
  cvt_f32_f16<<<(65536 + 255) / 256, 256, 0, stream>>>(valw, valw16, 65536);
  cvt_f32_f16<<<(2097152 + 255) / 256, 256, 0, stream>>>(finw, finw16, 2097152);

  // LayerNorm with in-LDS transpose: xn_T[b][l][c]
  const size_t ln_bytes = (size_t)256 * kTpad * sizeof(_Float16);  // 132 KB
  layernorm_k<<<kB * (kL / 256), 256, ln_bytes, stream>>>(x, g, be, xnT16);

  // fused logits + softmax: 128 KB logits tile in LDS (320 KB/WGP pool)
  const size_t lsm_bytes = (size_t)kQ * kL * sizeof(_Float16);  // 128 KB
  logits_softmax_k<<<kB * kHeads, 256, lsm_bytes, stream>>>(xnT16, attnw16, aw16);

  value_gemm_k<<<kB * 4 * 32, 256, 0, stream>>>(xnT16, valw16, valb, v16);
  pool_k<<<kB * kHeads / 8, 256, 0, stream>>>(aw16, v16, afl16);
  final_gemm_k<<<128 / 8, 256, 0, stream>>>(afl16, finw16, finb, out);
}
